// MultiHeadAttention_60833916780936
// MI455X (gfx1250) — matrix operand order repaired and checked by reading, not compiled, not run on hardware
//
#include <hip/hip_runtime.h>

// ---------------------------------------------------------------------------
// MHA forward for MI455X (gfx1250, wave32, WMMA bf16 16x16x32 + TDM staging)
// ---------------------------------------------------------------------------

typedef __bf16 bf16_t;
typedef __attribute__((ext_vector_type(16))) __bf16 v16bf;
typedef __attribute__((ext_vector_type(8)))  __bf16 v8bf;
typedef __attribute__((ext_vector_type(8)))  float  v8f;
typedef __attribute__((ext_vector_type(4)))  unsigned int v4u;
typedef __attribute__((ext_vector_type(8)))  int v8i;
typedef __attribute__((ext_vector_type(4)))  int v4i;

#define HNUM   16
#define DKV    64
#define DMODEL 1024
#define BB     2
#define LSEQ   2048
#define MROWS  (BB * LSEQ)          // 4096
#define TEMP_INV 0.125f             // 1/sqrt(64)
#define LN_EPS   1e-5f

#ifndef __has_builtin
#define __has_builtin(x) 0
#endif
#if __has_builtin(__builtin_amdgcn_tensor_load_to_lds) && \
    __has_builtin(__builtin_amdgcn_s_wait_tensorcnt)
#define HAVE_TDM 1
#else
#define HAVE_TDM 0
#endif

// ---------------- WMMA helpers -------------------------------------------

__device__ __forceinline__ v8f wmma_bf16(v16bf a, v16bf b, v8f c) {
  // (neg_a, A, neg_b, B, c_mod, C, reuse_a, reuse_b)
  return __builtin_amdgcn_wmma_f32_16x16x32_bf16(false, a, false, b, (short)0, c,
                                                 false, false);
}

// A fragment (16x32 bf16): lane r=lane&15 is row M=r; lanes<16 hold
// K={0..7,16..23}, lanes>=16 hold K={8..15,24..31}.
__device__ __forceinline__ v16bf load_frag_a(const bf16_t* tile, int stride, int kk) {
  const int lane = threadIdx.x & 31;
  const int r    = lane & 15;
  const int kb   = (lane >> 4) ? 8 : 0;
  const bf16_t* p = tile + r * stride + kk + kb;
  v8bf lo = *(const v8bf*)p;          // K = kb..kb+7
  v8bf hi = *(const v8bf*)(p + 16);   // K = kb+16..kb+23
  return __builtin_shufflevector(lo, hi, 0, 1, 2, 3, 4, 5, 6, 7,
                                 8, 9, 10, 11, 12, 13, 14, 15);
}

// B fragment (32x16 bf16) from LDS stored [n][k]: lane n=lane&15 is column N;
// lanes<16 hold K={0..7,16..23}, lanes>=16 hold K={8..15,24..31}.
__device__ __forceinline__ v16bf load_frag_b(const bf16_t* tile, int stride, int kk) {
  const int lane = threadIdx.x & 31;
  const int n    = lane & 15;
  const int kb   = (lane >> 4) ? 8 : 0;
  const bf16_t* p = tile + n * stride + kk + kb;
  v8bf lo = *(const v8bf*)p;
  v8bf hi = *(const v8bf*)(p + 16);
  return __builtin_shufflevector(lo, hi, 0, 1, 2, 3, 4, 5, 6, 7,
                                 8, 9, 10, 11, 12, 13, 14, 15);
}

// ---------------- staging helpers ----------------------------------------

__device__ __forceinline__ void cvt_store8(bf16_t* dst, float4 a, float4 b) {
  v8bf p;
  p[0] = (bf16_t)a.x; p[1] = (bf16_t)a.y; p[2] = (bf16_t)a.z; p[3] = (bf16_t)a.w;
  p[4] = (bf16_t)b.x; p[5] = (bf16_t)b.y; p[6] = (bf16_t)b.z; p[7] = (bf16_t)b.w;
  *(v8bf*)dst = p;
}

// 64x32 f32 tile -> LDS bf16 [64][48]; 128 threads, 16 elems each.
__device__ __forceinline__ void stage_f32_tile(bf16_t (*dst)[48],
                                               const float* __restrict__ src,
                                               int ldsrc) {
  const int row   = threadIdx.x & 63;
  const int cbase = (threadIdx.x >> 6) * 16;
  const float4* s = (const float4*)(src + (size_t)row * ldsrc + cbase);
  float4 f0 = s[0], f1 = s[1], f2 = s[2], f3 = s[3];
  cvt_store8(&dst[row][cbase], f0, f1);
  cvt_store8(&dst[row][cbase + 8], f2, f3);
}

// 64x32 bf16 tile copy -> LDS [64][48]
__device__ __forceinline__ void stage_bf16_tile32(bf16_t (*dst)[48],
                                                  const bf16_t* __restrict__ src,
                                                  int ldsrc) {
  const int row   = threadIdx.x & 63;
  const int cbase = (threadIdx.x >> 6) * 16;
  const uint4* s = (const uint4*)(src + (size_t)row * ldsrc + cbase);
  uint4* d = (uint4*)&dst[row][cbase];
  d[0] = s[0]; d[1] = s[1];
}

// 64x64 bf16 tile copy -> LDS [64][80] (manual fallback for TDM path)
__device__ __forceinline__ void stage_bf16_tile64(bf16_t (*dst)[80],
                                                  const bf16_t* __restrict__ src,
                                                  int ldsrc) {
  const int row   = threadIdx.x & 63;
  const int cbase = (threadIdx.x >> 6) * 32;
  const uint4* s = (const uint4*)(src + (size_t)row * ldsrc + cbase);
  uint4* d = (uint4*)&dst[row][cbase];
  d[0] = s[0]; d[1] = s[1]; d[2] = s[2]; d[3] = s[3];
}

// ---------------- Tensor Data Mover: 64x64 bf16 tile -> padded LDS --------
// Loads a 64-row x 64-col bf16 tile (row stride DMODEL elements) from a
// [4096 x 1024] bf16 tensor into LDS with +16-element row padding
// (pad_interval = 32 DWORDs per row, pad_amount = 8 DWORDs) so the LDS image
// is exactly the [64][80] layout the fragment loaders expect.
#if HAVE_TDM
__device__ __forceinline__ void tdm_load_tile_64x64(const bf16_t* gsrc,
                                                    void* lds_dst) {
  const unsigned long long ga = (unsigned long long)(uintptr_t)gsrc;
  v4u g0 = {};
  g0[0] = 1u;                                   // count = 1 valid descriptor
  g0[1] = (unsigned)(uintptr_t)lds_dst;         // lds_addr (byte offset)
  g0[2] = (unsigned)ga;                         // global_addr[31:0]
  g0[3] = (unsigned)((ga >> 32) & 0x01FFFFFFu)  // global_addr[56:32]
          | 0x80000000u;                        // type = 2 ("image")
  v8i g1 = {};
  g1[0] = (int)((1u << 16)      // data_size = 1 (2 bytes)
                | (1u << 20)    // pad_enable
                | (4u << 22)    // pad_interval: 32 DWORDs (= one 64-elem row)
                | (7u << 25));  // pad_amount: 8 DWORDs (= 16 bf16 elements)
  g1[1] = (int)((unsigned)DMODEL << 16);        // tensor_dim0[15:0] = 1024
  g1[2] = (int)((unsigned)MROWS << 16);         // tensor_dim1[15:0] = 4096
  g1[3] = (int)(64u << 16);                     // tile_dim0 = 64
  g1[4] = 64;                                   // tile_dim1 = 64
  g1[5] = DMODEL;                               // tensor_dim0_stride = 1024
  v4i z4 = {};
#if defined(__clang_major__) && (__clang_major__ >= 23)
  v8i z8 = {};
  __builtin_amdgcn_tensor_load_to_lds(g0, g1, z4, z4, z8, 0);
#else
  __builtin_amdgcn_tensor_load_to_lds(g0, g1, z4, z4, 0);
#endif
}
#endif

// ---------------- kernel 1: input projection ------------------------------
// Y[m][n] = sum_k X[m][k] * W[n][k] + bias[n]   (M=4096, N=1024, K=1024)
// Double-buffered LDS; writes Yb (bf16). Q projection also scatters q_flat.
__global__ __launch_bounds__(128)
void proj_gemm_kernel(const float* __restrict__ X, const float* __restrict__ W,
                      const float* __restrict__ bias, bf16_t* __restrict__ Yb,
                      float* __restrict__ qflat) {
  __shared__ bf16_t As[2][64][48];
  __shared__ bf16_t Bs[2][64][48];
  const int tid  = threadIdx.x;
  const int wave = tid >> 5;
  const int lane = tid & 31;
  const int m0 = blockIdx.y * 64;
  const int n0 = blockIdx.x * 64;
  v8f acc[4] = {};

  stage_f32_tile(As[0], X + (size_t)m0 * DMODEL, DMODEL);
  stage_f32_tile(Bs[0], W + (size_t)n0 * DMODEL, DMODEL);
  __syncthreads();

  for (int kk0 = 0; kk0 < DMODEL; kk0 += 32) {
    const int cur = (kk0 >> 5) & 1;
    if (kk0 + 32 < DMODEL) {
      stage_f32_tile(As[cur ^ 1], X + (size_t)m0 * DMODEL + kk0 + 32, DMODEL);
      stage_f32_tile(Bs[cur ^ 1], W + (size_t)n0 * DMODEL + kk0 + 32, DMODEL);
    }
    v16bf afrag = load_frag_a(&As[cur][wave * 16][0], 48, 0);
#pragma unroll
    for (int j = 0; j < 4; ++j) {
      v16bf bfrag = load_frag_b(&Bs[cur][j * 16][0], 48, 0);
      acc[j] = wmma_bf16(afrag, bfrag, acc[j]);
    }
    __syncthreads();
  }

  const int mrow = m0 + wave * 16 + ((lane >> 4) ? 8 : 0);
  const int ncol = lane & 15;
#pragma unroll
  for (int j = 0; j < 4; ++j) {
    const int gn = n0 + j * 16 + ncol;
    const float bv = bias[gn];
#pragma unroll
    for (int r = 0; r < 8; ++r) {
      const int gm = mrow + r;
      const float val = acc[j][r] + bv;
      Yb[(size_t)gm * DMODEL + gn] = (bf16_t)val;
      if (qflat) {
        const int b  = gm >> 11;        // / LSEQ
        const int lq = gm & (LSEQ - 1);
        const int h  = gn >> 6;         // / DKV
        const int d  = gn & (DKV - 1);
        qflat[(((size_t)(h * BB + b) * LSEQ) + lq) * DKV + d] = val;
      }
    }
  }
}

// ---------------- kernel 2: attention scores ------------------------------
// S[hb][i][j] = (qh[b,i,h,:] . kh[b,j,h,:]) / sqrt(dk)
// Tiles staged with the Tensor Data Mover (wave 0 issues the DMA, waits on
// TENSORcnt, then the workgroup barrier publishes LDS to all 4 waves).
__global__ __launch_bounds__(128)
void attn_scores_kernel(const bf16_t* __restrict__ Qb, const bf16_t* __restrict__ Kb,
                        float* __restrict__ attn) {
  __shared__ bf16_t Qs[64][80];
  __shared__ bf16_t Ks[64][80];
  const int tid  = threadIdx.x;
  const int wave = tid >> 5;
  const int lane = tid & 31;
  const int n0 = blockIdx.x * 64;
  const int m0 = blockIdx.y * 64;
  const int hb = blockIdx.z;        // h * B + b
  const int h = hb / BB, b = hb % BB;

  const bf16_t* qsrc = Qb + ((size_t)(b * LSEQ + m0)) * DMODEL + h * DKV;
  const bf16_t* ksrc = Kb + ((size_t)(b * LSEQ + n0)) * DMODEL + h * DKV;
#if HAVE_TDM
  if (wave == 0) {
    tdm_load_tile_64x64(qsrc, &Qs[0][0]);
    tdm_load_tile_64x64(ksrc, &Ks[0][0]);
    __builtin_amdgcn_s_wait_tensorcnt(0);
  }
#else
  stage_bf16_tile64(Qs, qsrc, DMODEL);
  stage_bf16_tile64(Ks, ksrc, DMODEL);
#endif
  __syncthreads();

  v8f acc[4] = {};
#pragma unroll
  for (int kk = 0; kk < DKV; kk += 32) {
    v16bf afrag = load_frag_a(&Qs[wave * 16][0], 80, kk);
#pragma unroll
    for (int j = 0; j < 4; ++j) {
      v16bf bfrag = load_frag_b(&Ks[j * 16][0], 80, kk);
      acc[j] = wmma_bf16(afrag, bfrag, acc[j]);
    }
  }

  float* out = attn + (size_t)hb * LSEQ * LSEQ;
  const int mrow = m0 + wave * 16 + ((lane >> 4) ? 8 : 0);
  const int ncol = lane & 15;
#pragma unroll
  for (int j = 0; j < 4; ++j) {
    const int gn = n0 + j * 16 + ncol;
#pragma unroll
    for (int r = 0; r < 8; ++r)
      out[(size_t)(mrow + r) * LSEQ + gn] = acc[j][r] * TEMP_INV;
  }
}

// ---------------- kernel 3: row softmax over LK ---------------------------
__global__ __launch_bounds__(256)
void softmax_kernel(float* __restrict__ attn) {
  __shared__ float red[8];
  float* p = attn + (size_t)blockIdx.x * LSEQ;
  const int tid = threadIdx.x;
  float4 v0 = ((float4*)p)[tid * 2];
  float4 v1 = ((float4*)p)[tid * 2 + 1];
  float m = fmaxf(fmaxf(fmaxf(v0.x, v0.y), fmaxf(v0.z, v0.w)),
                  fmaxf(fmaxf(v1.x, v1.y), fmaxf(v1.z, v1.w)));
#pragma unroll
  for (int off = 16; off; off >>= 1) m = fmaxf(m, __shfl_xor(m, off, 32));
  if ((tid & 31) == 0) red[tid >> 5] = m;
  __syncthreads();
  m = red[0];
#pragma unroll
  for (int i = 1; i < 8; ++i) m = fmaxf(m, red[i]);

  v0.x = __expf(v0.x - m); v0.y = __expf(v0.y - m);
  v0.z = __expf(v0.z - m); v0.w = __expf(v0.w - m);
  v1.x = __expf(v1.x - m); v1.y = __expf(v1.y - m);
  v1.z = __expf(v1.z - m); v1.w = __expf(v1.w - m);
  float s = v0.x + v0.y + v0.z + v0.w + v1.x + v1.y + v1.z + v1.w;
#pragma unroll
  for (int off = 16; off; off >>= 1) s += __shfl_xor(s, off, 32);
  __syncthreads();
  if ((tid & 31) == 0) red[tid >> 5] = s;
  __syncthreads();
  s = red[0] + red[1] + red[2] + red[3] + red[4] + red[5] + red[6] + red[7];
  const float inv = 1.0f / s;
  v0.x *= inv; v0.y *= inv; v0.z *= inv; v0.w *= inv;
  v1.x *= inv; v1.y *= inv; v1.z *= inv; v1.w *= inv;
  ((float4*)p)[tid * 2]     = v0;
  ((float4*)p)[tid * 2 + 1] = v1;
}

// ---------------- kernel 4: context = attn @ vh ---------------------------
// ctx[b][i][h*64+n] = sum_j attn[hb][i][j] * vh[b][j][h*64+n]
__global__ __launch_bounds__(128)
void attn_ctx_kernel(const float* __restrict__ attn, const bf16_t* __restrict__ Vb,
                     bf16_t* __restrict__ Cb) {
  __shared__ bf16_t As[2][64][48];
  __shared__ bf16_t Bs[2][64][48];   // stored [n][k] (transposed on stage-in)
  const int tid  = threadIdx.x;
  const int wave = tid >> 5;
  const int lane = tid & 31;
  const int m0 = blockIdx.x * 64;
  const int hb = blockIdx.y;
  const int h = hb / BB, b = hb % BB;
  const float* arow = attn + (size_t)hb * LSEQ * LSEQ;
  v8f acc[4] = {};

  const int jrow  = tid & 31;          // B staging: k-row within 32
  const int nbase = (tid >> 5) * 16;   // B staging: 16-column strip

  // stage first tiles into buffer 0
  stage_f32_tile(As[0], arow + (size_t)m0 * LSEQ, LSEQ);
  {
    const v8bf* vs = (const v8bf*)(Vb + ((size_t)(b * LSEQ + jrow)) * DMODEL +
                                   h * DKV + nbase);
    v8bf x0 = vs[0], x1 = vs[1];
#pragma unroll
    for (int c = 0; c < 8; ++c) Bs[0][nbase + c][jrow] = x0[c];
#pragma unroll
    for (int c = 0; c < 8; ++c) Bs[0][nbase + 8 + c][jrow] = x1[c];
  }
  __syncthreads();

  for (int k0 = 0; k0 < LSEQ; k0 += 32) {
    const int cur = (k0 >> 5) & 1;
    if (k0 + 32 < LSEQ) {
      stage_f32_tile(As[cur ^ 1], arow + (size_t)m0 * LSEQ + k0 + 32, LSEQ);
      const v8bf* vs = (const v8bf*)(Vb +
                        ((size_t)(b * LSEQ + k0 + 32 + jrow)) * DMODEL +
                        h * DKV + nbase);
      v8bf x0 = vs[0], x1 = vs[1];
#pragma unroll
      for (int c = 0; c < 8; ++c) Bs[cur ^ 1][nbase + c][jrow] = x0[c];
#pragma unroll
      for (int c = 0; c < 8; ++c) Bs[cur ^ 1][nbase + 8 + c][jrow] = x1[c];
    }
    // prefetch the attn tile two steps ahead into GL2 (global_prefetch_b8)
    if (k0 + 64 < LSEQ)
      __builtin_prefetch(arow + (size_t)(m0 + (tid & 63)) * LSEQ + k0 + 64, 0, 1);

    v16bf afrag = load_frag_a(&As[cur][wave * 16][0], 48, 0);
#pragma unroll
    for (int j = 0; j < 4; ++j) {
      v16bf bfrag = load_frag_b(&Bs[cur][j * 16][0], 48, 0);
      acc[j] = wmma_bf16(afrag, bfrag, acc[j]);
    }
    __syncthreads();
  }

  const int mrow = m0 + wave * 16 + ((lane >> 4) ? 8 : 0);
  const int ncol = lane & 15;
#pragma unroll
  for (int j = 0; j < 4; ++j) {
    const int gn = j * 16 + ncol;     // 0..63 within head
#pragma unroll
    for (int r = 0; r < 8; ++r)
      Cb[((size_t)(b * LSEQ + mrow + r)) * DMODEL + h * DKV + gn] = (bf16_t)acc[j][r];
  }
}

// ---------------- kernel 5: output projection + residual ------------------
// X[m][n] = sum_k ctx[m][k] * Wfc[n][k] + bfc[n] + resid[m][n]
__global__ __launch_bounds__(128)
void fc_residual_kernel(const bf16_t* __restrict__ Cb, const float* __restrict__ Wfc,
                        const float* __restrict__ bfc, const float* __restrict__ resid,
                        float* __restrict__ Xout) {
  __shared__ bf16_t As[2][64][48];
  __shared__ bf16_t Bs[2][64][48];
  const int tid  = threadIdx.x;
  const int wave = tid >> 5;
  const int lane = tid & 31;
  const int m0 = blockIdx.y * 64;
  const int n0 = blockIdx.x * 64;
  v8f acc[4] = {};

  stage_bf16_tile32(As[0], Cb + (size_t)m0 * DMODEL, DMODEL);
  stage_f32_tile(Bs[0], Wfc + (size_t)n0 * DMODEL, DMODEL);
  __syncthreads();

  for (int kk0 = 0; kk0 < DMODEL; kk0 += 32) {
    const int cur = (kk0 >> 5) & 1;
    if (kk0 + 32 < DMODEL) {
      stage_bf16_tile32(As[cur ^ 1], Cb + (size_t)m0 * DMODEL + kk0 + 32, DMODEL);
      stage_f32_tile(Bs[cur ^ 1], Wfc + (size_t)n0 * DMODEL + kk0 + 32, DMODEL);
    }
    v16bf afrag = load_frag_a(&As[cur][wave * 16][0], 48, 0);
#pragma unroll
    for (int j = 0; j < 4; ++j) {
      v16bf bfrag = load_frag_b(&Bs[cur][j * 16][0], 48, 0);
      acc[j] = wmma_bf16(afrag, bfrag, acc[j]);
    }
    __syncthreads();
  }

  const int mrow = m0 + wave * 16 + ((lane >> 4) ? 8 : 0);
  const int ncol = lane & 15;
#pragma unroll
  for (int j = 0; j < 4; ++j) {
    const int gn = n0 + j * 16 + ncol;
    const float bv = bfc[gn];
#pragma unroll
    for (int r = 0; r < 8; ++r) {
      const int gm = mrow + r;
      Xout[(size_t)gm * DMODEL + gn] =
          acc[j][r] + bv + resid[(size_t)gm * DMODEL + gn];
    }
  }
}

// ---------------- kernel 6: LayerNorm -------------------------------------
__global__ __launch_bounds__(256)
void layernorm_kernel(const float* __restrict__ X, const float* __restrict__ gamma,
                      const float* __restrict__ beta, float* __restrict__ Y) {
  __shared__ float red[8];
  const float* p = X + (size_t)blockIdx.x * DMODEL;
  const int tid = threadIdx.x;
  float4 v = ((const float4*)p)[tid];
  float s  = v.x + v.y + v.z + v.w;
  float sq = v.x * v.x + v.y * v.y + v.z * v.z + v.w * v.w;

#pragma unroll
  for (int off = 16; off; off >>= 1) s += __shfl_xor(s, off, 32);
  if ((tid & 31) == 0) red[tid >> 5] = s;
  __syncthreads();
  s = red[0] + red[1] + red[2] + red[3] + red[4] + red[5] + red[6] + red[7];
  __syncthreads();

#pragma unroll
  for (int off = 16; off; off >>= 1) sq += __shfl_xor(sq, off, 32);
  if ((tid & 31) == 0) red[tid >> 5] = sq;
  __syncthreads();
  sq = red[0] + red[1] + red[2] + red[3] + red[4] + red[5] + red[6] + red[7];

  const float mean = s * (1.0f / DMODEL);
  const float var  = sq * (1.0f / DMODEL) - mean * mean;
  const float rstd = rsqrtf(var + LN_EPS);
  float4 g = ((const float4*)gamma)[tid];
  float4 t = ((const float4*)beta)[tid];
  float4 o;
  o.x = (v.x - mean) * rstd * g.x + t.x;
  o.y = (v.y - mean) * rstd * g.y + t.y;
  o.z = (v.z - mean) * rstd * g.z + t.z;
  o.w = (v.w - mean) * rstd * g.w + t.w;
  ((float4*)(Y + (size_t)blockIdx.x * DMODEL))[tid] = o;
}

// ---------------- launcher -------------------------------------------------

extern "C" void kernel_launch(void* const* d_in, const int* in_sizes, int n_in,
                              void* d_out, int out_size, void* d_ws, size_t ws_size,
                              hipStream_t stream) {
  const float* q     = (const float*)d_in[0];
  const float* k     = (const float*)d_in[1];
  const float* v     = (const float*)d_in[2];
  const float* Wq    = (const float*)d_in[3];
  const float* bq    = (const float*)d_in[4];
  const float* Wk    = (const float*)d_in[5];
  const float* bk    = (const float*)d_in[6];
  const float* Wv    = (const float*)d_in[7];
  const float* bv    = (const float*)d_in[8];
  const float* Wfc   = (const float*)d_in[9];
  const float* bfc   = (const float*)d_in[10];
  const float* gamma = (const float*)d_in[11];
  const float* beta  = (const float*)d_in[12];

  float* out      = (float*)d_out;
  float* y_out    = out;                                    // [4096,1024]
  float* attn_out = out + (size_t)MROWS * DMODEL;           // [32,2048,2048]
  float* qflat    = attn_out + (size_t)HNUM * BB * LSEQ * LSEQ;  // [32,2048,64]

  char* ws = (char*)d_ws;
  bf16_t* qh = (bf16_t*)(ws + (size_t)(0u << 20));   // 8 MB each
  bf16_t* kh = (bf16_t*)(ws + (size_t)(8u << 20));
  bf16_t* vh = (bf16_t*)(ws + (size_t)(16u << 20));
  bf16_t* cb = (bf16_t*)(ws + (size_t)(24u << 20));
  float*  xb = (float*) (ws + (size_t)(32u << 20));  // 16 MB

  const dim3 gproj(DMODEL / 64, MROWS / 64);         // (16, 64)
  proj_gemm_kernel<<<gproj, 128, 0, stream>>>(q, Wq, bq, qh, qflat);
  proj_gemm_kernel<<<gproj, 128, 0, stream>>>(k, Wk, bk, kh, nullptr);
  proj_gemm_kernel<<<gproj, 128, 0, stream>>>(v, Wv, bv, vh, nullptr);

  const dim3 gs(LSEQ / 64, LSEQ / 64, HNUM * BB);    // (32, 32, 32)
  attn_scores_kernel<<<gs, 128, 0, stream>>>(qh, kh, attn_out);

  softmax_kernel<<<dim3(HNUM * BB * LSEQ), 256, 0, stream>>>(attn_out);

  attn_ctx_kernel<<<dim3(LSEQ / 64, HNUM * BB), 128, 0, stream>>>(attn_out, vh, cb);

  fc_residual_kernel<<<gproj, 128, 0, stream>>>(cb, Wfc, bfc, q, xb);

  layernorm_kernel<<<dim3(MROWS), 256, 0, stream>>>(xb, gamma, beta, y_out);
}
